// AttentionBlock_40750649705130
// MI455X (gfx1250) — matrix operand order repaired and checked by reading, not compiled, not run on hardware
//
#include <hip/hip_runtime.h>

// ---------------------------------------------------------------------------
// AttentionBlock for MI455X (gfx1250, wave32, WMMA).
// All GEMMs use v_wmma_f32_16x16x32_f16 (f16 A/B, f32 accum).
// Shapes: B=32, N=H*W=1024, C=256, 3C=768, GROUPS=32 (8 ch/group).
// ---------------------------------------------------------------------------

typedef __attribute__((ext_vector_type(4)))  _Float16 v4h;
typedef __attribute__((ext_vector_type(8)))  _Float16 v8h;
typedef __attribute__((ext_vector_type(16))) _Float16 v16h;
typedef __attribute__((ext_vector_type(4)))  float    v4f;
typedef __attribute__((ext_vector_type(8)))  float    v8f;

#define DEVINL __device__ __forceinline__

static constexpr int BATCH = 32;
static constexpr int NTOK  = 1024;   // H*W
static constexpr int CCH   = 256;
static constexpr int C3    = 768;

// Workspace layout (bytes). Requires ws_size >= ~219 MB.
static constexpr size_t H_OFF   = 0;                         // 16 MiB f16 [B*N, C]
static constexpr size_t Q_OFF   = H_OFF  + (size_t)BATCH*NTOK*CCH*2;   // f16 [B*N, C] (pre-scaled)
static constexpr size_t K_OFF   = Q_OFF  + (size_t)BATCH*NTOK*CCH*2;   // f16 [B*N, C]
static constexpr size_t VT_OFF  = K_OFF  + (size_t)BATCH*NTOK*CCH*2;   // f16 [B, C, N] (transposed)
static constexpr size_t O_OFF   = VT_OFF + (size_t)BATCH*NTOK*CCH*2;   // f16 [B*N, C]
static constexpr size_t S_OFF   = O_OFF  + (size_t)BATCH*NTOK*CCH*2;   // f32 [B*N, 1024] scores / f16 probs
static constexpr size_t WTQ_OFF = S_OFF  + (size_t)BATCH*NTOK*1024*4;  // f16 [768, 256] (w_qkv^T)
static constexpr size_t WTO_OFF = WTQ_OFF + (size_t)C3*CCH*2;          // f16 [256, 256] (w_out^T)

// ---------------------------------------------------------------------------
// WMMA fragment loaders — contiguous k order shared by A and B (not the ISA table order).
// A 16x32 f16: lane m=l%16 is row; g=l/16 selects K-half:
//   halves j -> K = kc + 16g + j
DEVINL v16h load_a_frag(const _Float16* __restrict__ base, int ld, int row0, int kc) {
  int lane = threadIdx.x & 31;
  int m = lane & 15, g = lane >> 4;
  const _Float16* p = base + (size_t)(row0 + m) * ld + kc + g * 16;
  v16h a = *(const v16h*)(p);
  return a;
}

// B 32x16 f16: lane n=l%16 is column; halves j -> K = kc + 16*(l/16) + j.
// 'base' stores each B-column as a contiguous K-major row of length ld.
DEVINL v16h load_b_frag(const _Float16* __restrict__ base, int ld, int col0, int kc) {
  int lane = threadIdx.x & 31;
  int n = lane & 15, g = lane >> 4;
  const _Float16* p = base + (size_t)(col0 + n) * ld + kc + g * 16;
  return *(const v16h*)p;
}

DEVINL v8f wmma_f16(v16h a, v16h b, v8f c) {
  return __builtin_amdgcn_wmma_f32_16x16x32_f16(false, a, false, b, (short)0, c,
                                                false, false);
}

// ---------------------------------------------------------------------------
// K0: weight prep — f32 -> f16, transposed so B-fragments are contiguous.
__global__ void k_prep(const float* __restrict__ w_qkv, const float* __restrict__ w_out,
                       _Float16* __restrict__ WTq, _Float16* __restrict__ WTo) {
  int idx = blockIdx.x * 256 + threadIdx.x;
  if (idx < C3 * CCH) {                       // WTq[col][k] = w_qkv[k][col]
    int col = idx >> 8, k = idx & 255;
    WTq[idx] = (_Float16)w_qkv[k * C3 + col];
  } else {
    int o = idx - C3 * CCH;                   // WTo[col][k] = w_out[k][col]
    int col = o >> 8, k = o & 255;
    WTo[o] = (_Float16)w_out[k * CCH + col];
  }
}

// ---------------------------------------------------------------------------
// K1: GroupNorm. One block per (batch, group); reduce 1024*8 elements.
__global__ void k_groupnorm(const float* __restrict__ x, const float* __restrict__ gamma,
                            const float* __restrict__ beta, _Float16* __restrict__ H) {
  __shared__ float s1[256];
  __shared__ float s2[256];
  int b = blockIdx.x >> 5, g = blockIdx.x & 31;
  int tid = threadIdx.x;
  const float* xb = x + (size_t)b * NTOK * CCH + g * 8;
  float sum = 0.f, sq = 0.f;
  for (int i = tid; i < 2048; i += 256) {     // 2048 float4 chunks
    int hw = i >> 1, off = (i & 1) * 4;
    v4f v = *(const v4f*)(xb + (size_t)hw * CCH + off);
#pragma unroll
    for (int j = 0; j < 4; ++j) { sum += v[j]; sq += v[j] * v[j]; }
  }
  s1[tid] = sum; s2[tid] = sq;
  __syncthreads();
  for (int s = 128; s > 0; s >>= 1) {
    if (tid < s) { s1[tid] += s1[tid + s]; s2[tid] += s2[tid + s]; }
    __syncthreads();
  }
  float mean = s1[0] * (1.0f / 8192.0f);
  float var  = s2[0] * (1.0f / 8192.0f) - mean * mean;
  float rstd = rsqrtf(var + 1e-3f);
  for (int i = tid; i < 2048; i += 256) {
    int hw = i >> 1, off = (i & 1) * 4;
    int c0 = g * 8 + off;
    v4f v = *(const v4f*)(xb + (size_t)hw * CCH + off);
    v4h h;
#pragma unroll
    for (int j = 0; j < 4; ++j)
      h[j] = (_Float16)((v[j] - mean) * rstd * gamma[c0 + j] + beta[c0 + j]);
    *(v4h*)(H + ((size_t)b * NTOK + hw) * CCH + c0) = h;
  }
}

// ---------------------------------------------------------------------------
// K2: QKV GEMM. [32768,256] x [256,768]. Split q|v|k, scale q, store v^T.
__global__ void k_qkv(const _Float16* __restrict__ H, const _Float16* __restrict__ WTq,
                      const float* __restrict__ b_qkv, const float* __restrict__ scale,
                      _Float16* __restrict__ Q, _Float16* __restrict__ Kb,
                      _Float16* __restrict__ VT) {
  int lane = threadIdx.x & 31, wave = threadIdx.x >> 5;
  int wm = wave & 3, wn = wave >> 2;
  int mBase = blockIdx.x * 128 + wm * 32;
  int nBase = blockIdx.y * 128 + wn * 64;
  v8f zero = {0.f, 0.f, 0.f, 0.f, 0.f, 0.f, 0.f, 0.f};
  v8f acc[2][4];
#pragma unroll
  for (int i = 0; i < 2; ++i)
#pragma unroll
    for (int j = 0; j < 4; ++j) acc[i][j] = zero;

  for (int kc = 0; kc < CCH; kc += 32) {
    v16h a0 = load_a_frag(H, CCH, mBase,      kc);
    v16h a1 = load_a_frag(H, CCH, mBase + 16, kc);
#pragma unroll
    for (int nj = 0; nj < 4; ++nj) {
      v16h b = load_b_frag(WTq, CCH, nBase + nj * 16, kc);
      acc[0][nj] = wmma_f16(a0, b, acc[0][nj]);
      acc[1][nj] = wmma_f16(a1, b, acc[1][nj]);
    }
  }
  float scl = scale[0];
  int n = lane & 15, g = lane >> 4;
#pragma unroll
  for (int mi = 0; mi < 2; ++mi) {
    int rowBase = mBase + mi * 16 + g * 8;
#pragma unroll
    for (int nj = 0; nj < 4; ++nj) {
      int col = nBase + nj * 16 + n;
      float bias = b_qkv[col];
      int sec = col >> 8, cc = col & 255;
      if (sec == 0) {                          // q, pre-scaled
#pragma unroll
        for (int e = 0; e < 8; ++e)
          Q[(size_t)(rowBase + e) * CCH + cc] = (_Float16)((acc[mi][nj][e] + bias) * scl);
      } else if (sec == 2) {                   // k
#pragma unroll
        for (int e = 0; e < 8; ++e)
          Kb[(size_t)(rowBase + e) * CCH + cc] = (_Float16)(acc[mi][nj][e] + bias);
      } else {                                 // v, stored transposed [b, c, token]
        int bb = rowBase >> 10, tok = rowBase & 1023;
        v8h h;
#pragma unroll
        for (int e = 0; e < 8; ++e) h[e] = (_Float16)(acc[mi][nj][e] + bias);
        *(v8h*)(VT + (((size_t)bb * CCH + cc) << 10) + tok) = h;
      }
    }
  }
}

// ---------------------------------------------------------------------------
// K3a: S = q @ k^T per batch. grid (8 mtile, 8 ntile, 32 batch).
__global__ void k_scores(const _Float16* __restrict__ Q, const _Float16* __restrict__ Kb,
                         float* __restrict__ S) {
  int lane = threadIdx.x & 31, wave = threadIdx.x >> 5;
  int wm = wave & 3, wn = wave >> 2;
  const _Float16* q = Q  + (size_t)blockIdx.z * NTOK * CCH;
  const _Float16* k = Kb + (size_t)blockIdx.z * NTOK * CCH;
  float* s = S + (size_t)blockIdx.z * NTOK * 1024;
  int mBase = blockIdx.x * 128 + wm * 32;
  int nBase = blockIdx.y * 128 + wn * 64;
  v8f zero = {0.f, 0.f, 0.f, 0.f, 0.f, 0.f, 0.f, 0.f};
  v8f acc[2][4];
#pragma unroll
  for (int i = 0; i < 2; ++i)
#pragma unroll
    for (int j = 0; j < 4; ++j) acc[i][j] = zero;

  for (int kc = 0; kc < CCH; kc += 32) {
    v16h a0 = load_a_frag(q, CCH, mBase,      kc);
    v16h a1 = load_a_frag(q, CCH, mBase + 16, kc);
#pragma unroll
    for (int nj = 0; nj < 4; ++nj) {
      v16h b = load_b_frag(k, CCH, nBase + nj * 16, kc);  // rows of k == cols of k^T
      acc[0][nj] = wmma_f16(a0, b, acc[0][nj]);
      acc[1][nj] = wmma_f16(a1, b, acc[1][nj]);
    }
  }
  int n = lane & 15, g = lane >> 4;
#pragma unroll
  for (int mi = 0; mi < 2; ++mi) {
    int rowBase = mBase + mi * 16 + g * 8;
#pragma unroll
    for (int nj = 0; nj < 4; ++nj) {
      int col = nBase + nj * 16 + n;
#pragma unroll
      for (int e = 0; e < 8; ++e)
        s[(size_t)(rowBase + e) * 1024 + col] = acc[mi][nj][e];
    }
  }
}

// ---------------------------------------------------------------------------
// K3b: row softmax; rewrite row in place as f16 probabilities (stride 2048 h).
__global__ void k_softmax(float* __restrict__ S) {
  __shared__ float sd[256];
  int tid = threadIdx.x;
  float* row = S + (size_t)blockIdx.x * 1024;
  v4f v = *(const v4f*)(row + tid * 4);       // all reads before any write
  float m = fmaxf(fmaxf(v[0], v[1]), fmaxf(v[2], v[3]));
  sd[tid] = m;
  __syncthreads();
  for (int s = 128; s > 0; s >>= 1) {
    if (tid < s) sd[tid] = fmaxf(sd[tid], sd[tid + s]);
    __syncthreads();
  }
  float rmax = sd[0];
  __syncthreads();
  float p0 = __expf(v[0] - rmax), p1 = __expf(v[1] - rmax);
  float p2 = __expf(v[2] - rmax), p3 = __expf(v[3] - rmax);
  sd[tid] = p0 + p1 + p2 + p3;
  __syncthreads();
  for (int s = 128; s > 0; s >>= 1) {
    if (tid < s) sd[tid] += sd[tid + s];
    __syncthreads();
  }
  float inv = 1.0f / sd[0];
  __syncthreads();
  v4h h;
  h[0] = (_Float16)(p0 * inv); h[1] = (_Float16)(p1 * inv);
  h[2] = (_Float16)(p2 * inv); h[3] = (_Float16)(p3 * inv);
  *(v4h*)((_Float16*)row + tid * 4) = h;      // f16 probs in first half of row
}

// ---------------------------------------------------------------------------
// K3c: O = P @ V per batch. P f16 (row stride 2048), V^T contiguous rows.
__global__ void k_attnv(const float* __restrict__ S, const _Float16* __restrict__ VT,
                        _Float16* __restrict__ O) {
  int lane = threadIdx.x & 31, wave = threadIdx.x >> 5;
  int wm = wave & 3, wn = wave >> 2;
  const _Float16* P  = (const _Float16*)(S + (size_t)blockIdx.z * NTOK * 1024);
  const _Float16* vt = VT + (size_t)blockIdx.z * CCH * NTOK;
  int mBase = blockIdx.x * 128 + wm * 32;
  int nBase = blockIdx.y * 128 + wn * 64;
  v8f zero = {0.f, 0.f, 0.f, 0.f, 0.f, 0.f, 0.f, 0.f};
  v8f acc[2][4];
#pragma unroll
  for (int i = 0; i < 2; ++i)
#pragma unroll
    for (int j = 0; j < 4; ++j) acc[i][j] = zero;

  for (int kc = 0; kc < NTOK; kc += 32) {
    v16h a0 = load_a_frag(P, 2048, mBase,      kc);
    v16h a1 = load_a_frag(P, 2048, mBase + 16, kc);
#pragma unroll
    for (int nj = 0; nj < 4; ++nj) {
      v16h b = load_b_frag(vt, 1024, nBase + nj * 16, kc);
      acc[0][nj] = wmma_f16(a0, b, acc[0][nj]);
      acc[1][nj] = wmma_f16(a1, b, acc[1][nj]);
    }
  }
  int n = lane & 15, g = lane >> 4;
#pragma unroll
  for (int mi = 0; mi < 2; ++mi) {
    int rowBase = mBase + mi * 16 + g * 8;
#pragma unroll
    for (int nj = 0; nj < 4; ++nj) {
      int col = nBase + nj * 16 + n;
#pragma unroll
      for (int e = 0; e < 8; ++e)
        O[((size_t)blockIdx.z * NTOK + rowBase + e) * CCH + col] = (_Float16)acc[mi][nj][e];
    }
  }
}

// ---------------------------------------------------------------------------
// K4: out = O @ w_out + b_out + x  (f32 result, residual add).
__global__ void k_out(const _Float16* __restrict__ O, const _Float16* __restrict__ WTo,
                      const float* __restrict__ b_out, const float* __restrict__ x,
                      float* __restrict__ out) {
  int lane = threadIdx.x & 31, wave = threadIdx.x >> 5;
  int wm = wave & 3, wn = wave >> 2;
  int mBase = blockIdx.x * 128 + wm * 32;
  int nBase = blockIdx.y * 128 + wn * 64;
  v8f zero = {0.f, 0.f, 0.f, 0.f, 0.f, 0.f, 0.f, 0.f};
  v8f acc[2][4];
#pragma unroll
  for (int i = 0; i < 2; ++i)
#pragma unroll
    for (int j = 0; j < 4; ++j) acc[i][j] = zero;

  for (int kc = 0; kc < CCH; kc += 32) {
    v16h a0 = load_a_frag(O, CCH, mBase,      kc);
    v16h a1 = load_a_frag(O, CCH, mBase + 16, kc);
#pragma unroll
    for (int nj = 0; nj < 4; ++nj) {
      v16h b = load_b_frag(WTo, CCH, nBase + nj * 16, kc);
      acc[0][nj] = wmma_f16(a0, b, acc[0][nj]);
      acc[1][nj] = wmma_f16(a1, b, acc[1][nj]);
    }
  }
  int n = lane & 15, g = lane >> 4;
#pragma unroll
  for (int mi = 0; mi < 2; ++mi) {
    int rowBase = mBase + mi * 16 + g * 8;
#pragma unroll
    for (int nj = 0; nj < 4; ++nj) {
      int col = nBase + nj * 16 + n;
      float bias = b_out[col];
#pragma unroll
      for (int e = 0; e < 8; ++e) {
        size_t idx = (size_t)(rowBase + e) * CCH + col;
        out[idx] = acc[mi][nj][e] + bias + x[idx];
      }
    }
  }
}

// ---------------------------------------------------------------------------
extern "C" void kernel_launch(void* const* d_in, const int* in_sizes, int n_in,
                              void* d_out, int out_size, void* d_ws, size_t ws_size,
                              hipStream_t stream) {
  (void)in_sizes; (void)n_in; (void)out_size; (void)ws_size;
  const float* x      = (const float*)d_in[0];
  const float* gamma  = (const float*)d_in[1];
  const float* beta   = (const float*)d_in[2];
  const float* w_qkv  = (const float*)d_in[3];
  const float* b_qkv  = (const float*)d_in[4];
  const float* w_out  = (const float*)d_in[5];
  const float* b_out  = (const float*)d_in[6];
  const float* scale  = (const float*)d_in[7];
  float* out = (float*)d_out;

  char* ws = (char*)d_ws;
  _Float16* H   = (_Float16*)(ws + H_OFF);
  _Float16* Q   = (_Float16*)(ws + Q_OFF);
  _Float16* Kb  = (_Float16*)(ws + K_OFF);
  _Float16* VT  = (_Float16*)(ws + VT_OFF);
  _Float16* O   = (_Float16*)(ws + O_OFF);
  float*    S   = (float*)   (ws + S_OFF);
  _Float16* WTq = (_Float16*)(ws + WTQ_OFF);
  _Float16* WTo = (_Float16*)(ws + WTO_OFF);

  // K0: weight transpose+convert (262144 elements)
  k_prep<<<dim3(1024), dim3(256), 0, stream>>>(w_qkv, w_out, WTq, WTo);
  // K1: GroupNorm: one block per (batch, group)
  k_groupnorm<<<dim3(BATCH * 32), dim3(256), 0, stream>>>(x, gamma, beta, H);
  // K2: QKV GEMM: 32768 x 768, 128x128 tiles
  k_qkv<<<dim3(256, 6), dim3(256), 0, stream>>>(H, WTq, b_qkv, scale, Q, Kb, VT);
  // K3a: scores per batch: 1024 x 1024, 128x128 tiles
  k_scores<<<dim3(8, 8, 32), dim3(256), 0, stream>>>(Q, Kb, S);
  // K3b: softmax, one block per row
  k_softmax<<<dim3(BATCH * NTOK), dim3(256), 0, stream>>>(S);
  // K3c: O = P @ V per batch: 1024 x 256
  k_attnv<<<dim3(8, 2, 32), dim3(256), 0, stream>>>(S, VT, O);
  // K4: out projection + bias + residual
  k_out<<<dim3(256, 2), dim3(256), 0, stream>>>(O, WTo, b_out, x, out);
}
